// diRNN_69741678953280
// MI455X (gfx1250) — matrix operand order repaired and checked by reading, not compiled, not run on hardware
//
#include <hip/hip_runtime.h>
#include <hip/hip_bf16.h>

// ---------------------------------------------------------------- constants
constexpr int kB  = 64;    // batch
constexpr int kNU = 64;    // input dim
constexpr int kNX = 512;   // state dim
constexpr int kNY = 32;    // output dim
constexpr int kL  = 2;     // layers
constexpr int kT  = 1024;  // sequence length

constexpr int kNWG   = 16;            // persistent workgroups (one per WGP)
constexpr int kNJT   = kNX / 16;      // 32 col tiles of state
constexpr int kNKT   = kNX / 32;      // 16 k tiles (K=32 per WMMA)
constexpr int kNKTU  = kNU / 32;      // 2 k tiles for the input GEMM
constexpr int kNPT   = kNY / 16;      // 2 output col tiles

typedef __bf16 bf16;
typedef __attribute__((ext_vector_type(16))) __bf16 v16bf;
typedef __attribute__((ext_vector_type(8)))  float  v8f;

// ---------------------------------------------------------------- WMMA core
__device__ __forceinline__ v8f wmma_bf16(v16bf a, v16bf b, v8f c) {
  // D(16x16,f32) = A(16x32,bf16) * B(32x16,bf16) + C
  return __builtin_amdgcn_wmma_f32_16x16x32_bf16(
      /*neg_a=*/false, a, /*neg_b=*/false, b,
      /*c_mod=*/(short)0, c, /*reuse_a=*/false, /*reuse_b=*/false);
}

// A fragment (16x32), contiguous K order (matches packed B):
//   lane<16 : M=lane,    K = {k0..k0+15}
//   lane>=16: M=lane-16, K = {k0+16..k0+31}
// Row-major bf16 source: one contiguous 16-element run per lane.
__device__ __forceinline__ v16bf load_a_bf16(const bf16* base, int ld,
                                             int m0, int k0, int lane) {
  const int m  = m0 + (lane & 15);
  const int kA = k0 + ((lane >> 4) << 4);
  const bf16* p = base + m * ld + kA;
  v16bf a;
#pragma unroll
  for (int i = 0; i < 8; ++i) {
    a[i]     = p[i];
    a[i + 8] = p[i + 8];
  }
  return a;
}

// B fragments are pre-packed: tile = 32 lanes x 16 contiguous bf16 (32B/lane).
__device__ __forceinline__ v16bf load_b_packed(const bf16* tileBase, int lane) {
  return *(const v16bf*)(tileBase + lane * 16);
}

// ---------------------------------------------------------------- global barrier
__device__ __forceinline__ void grid_barrier(unsigned* counter, unsigned nwg,
                                             unsigned& epoch) {
  __threadfence();
  __syncthreads();
  if (threadIdx.x == 0) {
    epoch += 1;
    atomicAdd(counter, 1u);
    while (__hip_atomic_load(counter, __ATOMIC_RELAXED,
                             __HIP_MEMORY_SCOPE_AGENT) < epoch * nwg) {
      __builtin_amdgcn_s_sleep(1);
    }
  }
  __syncthreads();
  __threadfence();
}

// ---------------------------------------------------------------- setup kernels
// Gauss-Jordan inversion of E (L blocks). E ~ I + small jitter: no pivoting.
__global__ void invert_kernel(const float* __restrict__ E, float* __restrict__ A,
                              float* __restrict__ Inv) {
  const int l = blockIdx.x;
  const int n = kNX;
  float* Al = A + (size_t)l * n * n;
  float* Il = Inv + (size_t)l * n * n;
  for (int i = threadIdx.x; i < n * n; i += blockDim.x) {
    Al[i] = E[(size_t)l * n * n + i];
    Il[i] = ((i / n) == (i % n)) ? 1.0f : 0.0f;
  }
  __syncthreads();

  __shared__ float colp[kNX];
  __shared__ float invp;
  for (int p = 0; p < n; ++p) {
    if (threadIdx.x == 0) invp = 1.0f / Al[p * n + p];
    __syncthreads();
    const float ip = invp;
    for (int c = threadIdx.x; c < n; c += blockDim.x) {
      Al[p * n + c] *= ip;
      Il[p * n + c] *= ip;
    }
    __syncthreads();
    for (int r = threadIdx.x; r < n; r += blockDim.x)
      colp[r] = (r == p) ? 0.0f : Al[r * n + p];
    __syncthreads();
    for (int idx = threadIdx.x; idx < n * n; idx += blockDim.x) {
      const int r = idx >> 9, c = idx & 511;
      const float f = colp[r];
      if (f != 0.0f) {
        Al[idx] -= f * Al[p * n + c];
        Il[idx] -= f * Il[p * n + c];
      }
    }
    __syncthreads();
  }
}

// Pack HwT B-fragments: layout [l][jt(32)][kt(16)][lane(32)][i(16)]
__global__ void pack_hw(const float* __restrict__ Hw, bf16* __restrict__ dst) {
  const int gid = blockIdx.x * blockDim.x + threadIdx.x;
  if (gid >= kL * kNJT * kNKT * 512) return;
  const int i = gid & 15, lane = (gid >> 4) & 31;
  const int kt = (gid >> 9) & 15, jt = (gid >> 13) & 31, l = gid >> 18;
  const int n = jt * 16 + (lane & 15);
  const int k = kt * 32 + ((lane >> 4) << 4) + i;
  dst[gid] = (__bf16)Hw[((size_t)l * kNX + n) * kNX + k];  // B[k][n] = Hw[n][k]
}

__global__ void pack_einv(const float* __restrict__ Einv, bf16* __restrict__ dst) {
  const int gid = blockIdx.x * blockDim.x + threadIdx.x;
  if (gid >= kL * kNJT * kNKT * 512) return;
  const int i = gid & 15, lane = (gid >> 4) & 31;
  const int kt = (gid >> 9) & 15, jt = (gid >> 13) & 31, l = gid >> 18;
  const int n = jt * 16 + (lane & 15);
  const int k = kt * 32 + ((lane >> 4) << 4) + i;
  dst[gid] = (__bf16)Einv[((size_t)l * kNX + k) * kNX + n];  // B[k][n] = Einv[k][n]
}

__global__ void pack_kw(const float* __restrict__ Kw, bf16* __restrict__ dst) {
  const int gid = blockIdx.x * blockDim.x + threadIdx.x;
  if (gid >= kL * kNJT * kNKTU * 512) return;
  const int i = gid & 15, lane = (gid >> 4) & 31;
  const int kt = (gid >> 9) & 1, jt = (gid >> 10) & 31, l = gid >> 15;
  const int n = jt * 16 + (lane & 15);
  const int k = kt * 32 + ((lane >> 4) << 4) + i;
  dst[gid] = (__bf16)Kw[((size_t)l * kNX + n) * kNU + k];  // B[k][n] = Kw[n][k]
}

__global__ void pack_cw(const float* __restrict__ Cw, bf16* __restrict__ dst) {
  const int gid = blockIdx.x * blockDim.x + threadIdx.x;
  if (gid >= kNPT * kNKT * 512) return;
  const int i = gid & 15, lane = (gid >> 4) & 31;
  const int kt = (gid >> 9) & 15, pt = gid >> 13;
  const int p = pt * 16 + (lane & 15);
  const int k = kt * 32 + ((lane >> 4) << 4) + i;
  dst[gid] = (__bf16)Cw[(size_t)p * kNX + k];  // B[k][p] = Cw[p][k]
}

// u (B,NU,T) f32 -> ubf (T,B,NU) bf16
__global__ void pack_u(const float* __restrict__ u, bf16* __restrict__ ubf) {
  const int gid = blockIdx.x * blockDim.x + threadIdx.x;
  if (gid >= kT * kB * kNU) return;
  const int nu = gid & 63, b = (gid >> 6) & 63, t = gid >> 12;
  ubf[gid] = (__bf16)u[((size_t)b * kNU + nu) * kT + t];
}

__global__ void init_kernel(bf16* __restrict__ Hbuf, unsigned* __restrict__ bar,
                            float* __restrict__ out, const float* __restrict__ Cb) {
  const int gid = blockIdx.x * blockDim.x + threadIdx.x;
  if (gid < kB * kNX) Hbuf[gid] = (__bf16)0.0f;      // h0 = 0
  if (gid == 0) *bar = 0u;                           // barrier epoch counter
  if (gid < kB * kNY) {                              // y[:, :, 0] = Cb (h0 == 0)
    const int b = gid >> 5, p = gid & 31;
    out[(size_t)b * kNY * kT + (size_t)p * kT + 0] = Cb[p];
  }
}

// ---------------------------------------------------------------- main scan
// Persistent kernel: 16 WGs x 256 threads (8 wave32s). Wave w of WG g owns the
// 16x16 output tile (mT = w&3, jT = 2g + (w>>2)) of every GEMM in the scan.
// State exchange buffers Hbuf/Xbuf are bf16 row-major: the WMMA consumes bf16
// anyway, so rounding once at D-store is numerically identical to converting
// at every A-load, but removes all cvt VALU work from the K-loops and halves
// the L2 exchange traffic.
__global__ void __launch_bounds__(256)
rnn_scan_kernel(const bf16* __restrict__ ubf, const float* __restrict__ Hb,
                const bf16* __restrict__ pHw, const bf16* __restrict__ pEinv,
                const bf16* __restrict__ pKw, const bf16* __restrict__ pCw,
                const float* __restrict__ Cb, bf16* __restrict__ Hbuf,
                bf16* __restrict__ Xbuf, unsigned* __restrict__ bar,
                float* __restrict__ out) {
  const int wg   = blockIdx.x;           // 0..15
  const int wave = threadIdx.x >> 5;     // 0..7  (wave32)
  const int lane = threadIdx.x & 31;

  const int mT = wave & 3;               // row tile 0..3  (M = 64)
  const int jT = wg * 2 + (wave >> 2);   // col tile 0..31 (N = 512)
  const int m0 = mT * 16;
  const int n0 = jT * 16;

  const int mrow = m0 + ((lane >> 4) << 3);  // D-frag: M = r + 8*(lane/16)
  const int ncol = n0 + (lane & 15);         // D-frag: N = lane%16
  unsigned epoch = 0;

  for (int t = 0; t < kT - 1; ++t) {
    const bf16* ut = ubf + (size_t)t * kB * kNU;
    for (int l = 0; l < kL; ++l) {
      // ---- stage 1: X = relu(H @ HwT + Hb + Ut @ KwT) -------------------
      v8f acc = {0.f, 0.f, 0.f, 0.f, 0.f, 0.f, 0.f, 0.f};
      const bf16* hwBase = pHw + (((size_t)(l * kNJT + jT)) * kNKT << 9);
      for (int kt = 0; kt < kNKT; ++kt) {
        if (kt + 1 < kNKT)
          __builtin_prefetch(hwBase + ((kt + 1) << 9) + lane * 16, 0, 3);
        v16bf a = load_a_bf16(Hbuf, kNX, m0, kt * 32, lane);
        v16bf b = load_b_packed(hwBase + (kt << 9), lane);
        acc = wmma_bf16(a, b, acc);
      }
      const bf16* kwBase = pKw + (((size_t)(l * kNJT + jT)) * kNKTU << 9);
      for (int kt = 0; kt < kNKTU; ++kt) {
        v16bf a = load_a_bf16(ut, kNU, m0, kt * 32, lane);
        v16bf b = load_b_packed(kwBase + (kt << 9), lane);
        acc = wmma_bf16(a, b, acc);
      }
      const float bias = Hb[l * kNX + ncol];
#pragma unroll
      for (int r = 0; r < 8; ++r)
        Xbuf[(size_t)(mrow + r) * kNX + ncol] =
            (__bf16)fmaxf(acc[r] + bias, 0.0f);
      grid_barrier(bar, kNWG, epoch);

      // ---- stage 2: H = X @ Einv ---------------------------------------
      v8f acc2 = {0.f, 0.f, 0.f, 0.f, 0.f, 0.f, 0.f, 0.f};
      const bf16* eiBase = pEinv + (((size_t)(l * kNJT + jT)) * kNKT << 9);
      for (int kt = 0; kt < kNKT; ++kt) {
        if (kt + 1 < kNKT)
          __builtin_prefetch(eiBase + ((kt + 1) << 9) + lane * 16, 0, 3);
        v16bf a = load_a_bf16(Xbuf, kNX, m0, kt * 32, lane);
        v16bf b = load_b_packed(eiBase + (kt << 9), lane);
        acc2 = wmma_bf16(a, b, acc2);
      }
#pragma unroll
      for (int r = 0; r < 8; ++r)
        Hbuf[(size_t)(mrow + r) * kNX + ncol] = (__bf16)acc2[r];
      grid_barrier(bar, kNWG, epoch);
    }

    // ---- output projection: y[:, :, t+1] = H @ CwT + Cb (WG 0 only) ----
    // Safe: the next write to Hbuf (stage 2 of t+1) is after a barrier that
    // WG0 joins only once it finished reading Hbuf here.
    if (wg == 0) {
      const int pT = wave >> 2;          // 0..1, with mT = wave & 3
      const int p0 = pT * 16;
      v8f acc = {0.f, 0.f, 0.f, 0.f, 0.f, 0.f, 0.f, 0.f};
      const bf16* cwBase = pCw + ((size_t)pT * kNKT << 9);
      for (int kt = 0; kt < kNKT; ++kt) {
        v16bf a = load_a_bf16(Hbuf, kNX, m0, kt * 32, lane);
        v16bf b = load_b_packed(cwBase + (kt << 9), lane);
        acc = wmma_bf16(a, b, acc);
      }
      const int p = p0 + (lane & 15);
      const float cb = Cb[p];
#pragma unroll
      for (int r = 0; r < 8; ++r) {
        const int bidx = m0 + ((lane >> 4) << 3) + r;
        out[(size_t)bidx * kNY * kT + (size_t)p * kT + (t + 1)] = acc[r] + cb;
      }
    }
  }
}

// ---------------------------------------------------------------- host glue
extern "C" void kernel_launch(void* const* d_in, const int* in_sizes, int n_in,
                              void* d_out, int out_size, void* d_ws, size_t ws_size,
                              hipStream_t stream) {
  (void)in_sizes; (void)n_in; (void)out_size; (void)ws_size;
  const float* u  = (const float*)d_in[0];  // (B, NU, T)
  const float* E  = (const float*)d_in[1];  // (L, NX, NX)
  const float* Kw = (const float*)d_in[2];  // (L, NX, NU)
  const float* Hw = (const float*)d_in[3];  // (L, NX, NX)
  const float* Hb = (const float*)d_in[4];  // (L, NX)
  const float* Cw = (const float*)d_in[5];  // (NY, NX)
  const float* Cb = (const float*)d_in[6];  // (NY,)
  float* out = (float*)d_out;

  char* ws = (char*)d_ws;
  size_t off = 0;
  auto take = [&](size_t bytes) -> char* {
    char* p = ws + off;
    off = (off + bytes + 255) & ~(size_t)255;
    return p;
  };
  float* Acopy = (float*)take((size_t)kL * kNX * kNX * sizeof(float));
  float* EinvF = (float*)take((size_t)kL * kNX * kNX * sizeof(float));
  bf16*  pHw   = (bf16*)take((size_t)kL * kNJT * kNKT * 512 * sizeof(bf16));
  bf16*  pEi   = (bf16*)take((size_t)kL * kNJT * kNKT * 512 * sizeof(bf16));
  bf16*  pKw   = (bf16*)take((size_t)kL * kNJT * kNKTU * 512 * sizeof(bf16));
  bf16*  pCw   = (bf16*)take((size_t)kNPT * kNKT * 512 * sizeof(bf16));
  bf16*  ubf   = (bf16*)take((size_t)kT * kB * kNU * sizeof(bf16));
  bf16*  Hbuf  = (bf16*)take((size_t)kB * kNX * sizeof(bf16));
  bf16*  Xbuf  = (bf16*)take((size_t)kB * kNX * sizeof(bf16));
  unsigned* bar = (unsigned*)take(sizeof(unsigned));

  // 1) Einv = inv(E)  (one WG per layer, Gauss-Jordan)
  invert_kernel<<<kL, 1024, 0, stream>>>(E, Acopy, EinvF);
  // 2) pack weights into per-lane WMMA B-fragment layouts (bf16)
  {
    const int nHw = kL * kNJT * kNKT * 512;
    pack_hw<<<(nHw + 255) / 256, 256, 0, stream>>>(Hw, pHw);
    pack_einv<<<(nHw + 255) / 256, 256, 0, stream>>>(EinvF, pEi);
    const int nKw = kL * kNJT * kNKTU * 512;
    pack_kw<<<(nKw + 255) / 256, 256, 0, stream>>>(Kw, pKw);
    const int nCw = kNPT * kNKT * 512;
    pack_cw<<<(nCw + 255) / 256, 256, 0, stream>>>(Cw, pCw);
    const int nU = kT * kB * kNU;
    pack_u<<<(nU + 255) / 256, 256, 0, stream>>>(u, ubf);
  }
  // 3) init state / barrier / y[:, :, 0]  (re-run every launch: replay-safe)
  init_kernel<<<(kB * kNX + 255) / 256, 256, 0, stream>>>(Hbuf, bar, out, Cb);
  // 4) persistent WMMA scan over time
  rnn_scan_kernel<<<kNWG, 256, 0, stream>>>(ubf, Hb, pHw, pEi, pKw, pCw, Cb,
                                            Hbuf, Xbuf, bar, out);
}
